// RGL_4810363372827
// MI455X (gfx1250) — matrix operand order repaired and checked by reading, not compiled, not run on hardware
//
#include <hip/hip_runtime.h>
#include <hip/hip_bf16.h>

typedef __attribute__((ext_vector_type(16))) _Float16 v16h;
typedef __attribute__((ext_vector_type(8)))  _Float16 v8h;
typedef __attribute__((ext_vector_type(8)))  float    v8f;

// ---- LDS layout (bytes). All f16 buffers use row strides that are multiples
// of 8 elements (16 B) so vector fragment loads are 16B-aligned.
#define OFF_A16   0u        // 128 x 136 f16 : h1, later attention A      (34816)
#define OFF_X16   34816u    // 128 x 72  f16 : X embeddings               (18432)
#define OFF_Y16   53248u    // 128 x 72  f16 : Y = X*w_a, later T = A*H   (18432)
#define OFF_H32   71680u    // 128 x 68  f32 : H (skip), earlier xin stage(34816)
#define OFF_HT16  106496u   //  64 x 136 f16 : H transposed (B operand)   (17408)
#define OFF_WB    123904u   // weight staging buffer                      (18432)
#define SMEM_BYTES 142336u

static __device__ __forceinline__ v16h pack16(v8h lo, v8h hi) {
  union { v16h v; v8h h[2]; } u;
  u.h[0] = lo; u.h[1] = hi;
  return u.v;
}

// A fragment (16x32 f16): lane row = m0+(lane&15); elems i -> K=k0+2*koffA+i,
// koffA = (lane>=16)?8:0.
static __device__ __forceinline__ v16h ldA(const _Float16* rowp, int k0, int koffA) {
  v8h lo = *(const v8h*)(rowp + k0 + 2 * koffA);
  v8h hi = *(const v8h*)(rowp + k0 + 2 * koffA + 8);
  return pack16(lo, hi);
}
// B fragment (32x16 f16): lane col = nt*16+(lane&15); elems i -> K=k0+koffB+i,
// koffB = (lane>=16)?16:0. rowp points at the (transposed) column's row.
static __device__ __forceinline__ v16h ldB(const _Float16* rowp, int k0, int koffB) {
  v8h lo = *(const v8h*)(rowp + k0 + koffB);
  v8h hi = *(const v8h*)(rowp + k0 + koffB + 8);
  return pack16(lo, hi);
}

static __device__ __forceinline__ v8f wmma_f16(v16h a, v16h b, v8f c) {
  return __builtin_amdgcn_wmma_f32_16x16x32_f16(false, a, false, b, (short)0, c,
                                                false, false);
}

__global__ __launch_bounds__(256)
void gcn_fused_kernel(const float* __restrict__ robot,   // [B,1,9]
                      const float* __restrict__ human,   // [B,127,5]
                      const float* __restrict__ wr_w1, const float* __restrict__ wr_b1,
                      const float* __restrict__ wr_w2, const float* __restrict__ wr_b2,
                      const float* __restrict__ wh_w1, const float* __restrict__ wh_b1,
                      const float* __restrict__ wh_w2, const float* __restrict__ wh_b2,
                      const float* __restrict__ w_a,
                      const float* __restrict__ wv_w, const float* __restrict__ wv_b,
                      const float* __restrict__ W0, const float* __restrict__ W1,
                      float* __restrict__ out)           // [B,128,64]
{
  extern __shared__ char smem[];
  _Float16* A16  = (_Float16*)(smem + OFF_A16);
  _Float16* X16  = (_Float16*)(smem + OFF_X16);
  _Float16* Y16  = (_Float16*)(smem + OFF_Y16);
  float*    H32  = (float*)   (smem + OFF_H32);
  _Float16* HT16 = (_Float16*)(smem + OFF_HT16);

  const int tid  = threadIdx.x;
  const int b    = blockIdx.x;
  const int lane = tid & 31;
  const int wv   = tid >> 5;          // wave id 0..7
  const int l16  = lane & 15;
  const int half = lane >> 4;
  const int koffA = half * 8;
  const int koffB = half * 16;
  const int m0    = wv * 16;          // this wave's M-tile rows

  // ================= Phase 0: stage inputs + layer-1 weights =================
  float* xin = H32;                   // reuse H32 region: [128][16] f32
  float* wbf = (float*)(smem + OFF_WB);
  float* wr1 = wbf;                   // 9*128
  float* wh1 = wbf + 1152;            // 5*128
  float* b1r = wbf + 1792;            // 128
  float* b1h = wbf + 1920;            // 128
  for (int i = tid; i < 1152; i += 256) wr1[i] = wr_w1[i];
  for (int i = tid; i < 640;  i += 256) wh1[i] = wh_w1[i];
  for (int i = tid; i < 128;  i += 256) { b1r[i] = wr_b1[i]; b1h[i] = wh_b1[i]; }
  for (int i = tid; i < 2048; i += 256) {
    int row = i >> 4, c = i & 15;
    float v = 0.f;
    if (row == 0) { if (c < 9) v = robot[b * 9 + c]; }
    else if (c < 5) v = human[((size_t)b * 127 + (row - 1)) * 5 + c];
    xin[i] = v;
  }
  __syncthreads();

  // ================= Phase 1: h1 = relu(x @ w1 + b1), f16 in A16 =============
  for (int it = 0; it < 64; ++it) {
    int idx = it * 256 + tid;
    int row = idx >> 7, j = idx & 127;
    float acc;
    if (row == 0) {
      acc = b1r[j];
#pragma unroll
      for (int k = 0; k < 9; ++k) acc += xin[k] * wr1[k * 128 + j];
    } else {
      acc = b1h[j];
#pragma unroll
      for (int k = 0; k < 5; ++k) acc += xin[row * 16 + k] * wh1[k * 128 + j];
    }
    A16[row * 136 + j] = (_Float16)fmaxf(acc, 0.f);
  }
  __syncthreads();

  // ================= Phase 2: X = relu(h1 @ wh_w2 + b2) via WMMA =============
  // K outermost: each A fragment is loaded once and reused for all 4 N-tiles.
  _Float16* wT136 = (_Float16*)(smem + OFF_WB); // w2^T : [64][136]
  for (int i = tid; i < 8192; i += 256) {
    int k = i >> 6, n = i & 63;
    wT136[n * 136 + k] = (_Float16)wh_w2[i];
  }
  __syncthreads();
  {
    const _Float16* Arow = A16 + (m0 + l16) * 136;
    v8f acc[4] = {};
#pragma unroll
    for (int k0 = 0; k0 < 128; k0 += 32) {
      v16h a = ldA(Arow, k0, koffA);
#pragma unroll
      for (int nt = 0; nt < 4; ++nt)
        acc[nt] = wmma_f16(a, ldB(wT136 + (nt * 16 + l16) * 136, k0, koffB), acc[nt]);
    }
#pragma unroll
    for (int nt = 0; nt < 4; ++nt) {
      int n = nt * 16 + l16;
      float bias = wh_b2[n];
#pragma unroll
      for (int r = 0; r < 8; ++r) {
        int row = m0 + r + half * 8;
        X16[row * 72 + n] = (_Float16)fmaxf(acc[nt][r] + bias, 0.f);
      }
    }
  }
  __syncthreads();
  // robot row uses wr_w2/wr_b2: overwrite row 0 with VALU (h1 row 0 used wr_w1)
  if (tid < 64) {
    float acc = wr_b2[tid];
    for (int k = 0; k < 128; ++k) acc += (float)A16[k] * wr_w2[k * 64 + tid];
    X16[tid] = (_Float16)fmaxf(acc, 0.f);
  }
  __syncthreads();

  // ================= Phase 3: Y = X @ w_a via WMMA ===========================
  _Float16* wT72 = (_Float16*)(smem + OFF_WB);  // [64][72] transposed weights
  for (int i = tid; i < 4096; i += 256) {
    int k = i >> 6, n = i & 63;
    wT72[n * 72 + k] = (_Float16)w_a[i];
  }
  __syncthreads();
  const _Float16* Xrow = X16 + (m0 + l16) * 72;
  {
    v8f acc[4] = {};
#pragma unroll
    for (int k0 = 0; k0 < 64; k0 += 32) {
      v16h a = ldA(Xrow, k0, koffA);
#pragma unroll
      for (int nt = 0; nt < 4; ++nt)
        acc[nt] = wmma_f16(a, ldB(wT72 + (nt * 16 + l16) * 72, k0, koffB), acc[nt]);
    }
#pragma unroll
    for (int nt = 0; nt < 4; ++nt) {
      int n = nt * 16 + l16;
#pragma unroll
      for (int r = 0; r < 8; ++r)
        Y16[(m0 + r + half * 8) * 72 + n] = (_Float16)acc[nt][r];
    }
  }
  __syncthreads();

  // ===== Phase 4: S = Y @ X^T (in regs), row softmax -> A f16 (reuse A16) ====
  {
    const _Float16* Yrow = Y16 + (m0 + l16) * 72;
    v8f S[8];
#pragma unroll
    for (int nt = 0; nt < 8; ++nt) S[nt] = v8f{};
#pragma unroll
    for (int k0 = 0; k0 < 64; k0 += 32) {
      v16h a = ldA(Yrow, k0, koffA);
#pragma unroll
      for (int nt = 0; nt < 8; ++nt) {
        int m = nt * 16 + l16;               // column = agent m; B from X rows
        S[nt] = wmma_f16(a, ldB(X16 + m * 72, k0, koffB), S[nt]);
      }
    }
#pragma unroll
    for (int r = 0; r < 8; ++r) {
      float mx = S[0][r];
#pragma unroll
      for (int nt = 1; nt < 8; ++nt) mx = fmaxf(mx, S[nt][r]);
#pragma unroll
      for (int msk = 1; msk < 16; msk <<= 1) mx = fmaxf(mx, __shfl_xor(mx, msk, 32));
      float e[8], sum = 0.f;
#pragma unroll
      for (int nt = 0; nt < 8; ++nt) { e[nt] = __expf(S[nt][r] - mx); sum += e[nt]; }
#pragma unroll
      for (int msk = 1; msk < 16; msk <<= 1) sum += __shfl_xor(sum, msk, 32);
      float inv = 1.f / sum;
      int row = m0 + r + half * 8;
#pragma unroll
      for (int nt = 0; nt < 8; ++nt)
        A16[row * 136 + nt * 16 + l16] = (_Float16)(e[nt] * inv);
    }
  }
  __syncthreads();

  // ============ Phase 5: H = relu(X @ wv_w + wv_b) -> H32 + HT16 =============
  for (int i = tid; i < 4096; i += 256) {
    int k = i >> 6, n = i & 63;
    wT72[n * 72 + k] = (_Float16)wv_w[i];
  }
  __syncthreads();
  {
    v8f acc[4] = {};
#pragma unroll
    for (int k0 = 0; k0 < 64; k0 += 32) {
      v16h a = ldA(Xrow, k0, koffA);
#pragma unroll
      for (int nt = 0; nt < 4; ++nt)
        acc[nt] = wmma_f16(a, ldB(wT72 + (nt * 16 + l16) * 72, k0, koffB), acc[nt]);
    }
#pragma unroll
    for (int nt = 0; nt < 4; ++nt) {
      int n = nt * 16 + l16;
      float bias = wv_b[n];
#pragma unroll
      for (int r = 0; r < 8; ++r) {
        int row = m0 + r + half * 8;
        float h = fmaxf(acc[nt][r] + bias, 0.f);
        H32[row * 68 + n] = h;
        HT16[n * 136 + row] = (_Float16)h;
      }
    }
  }
  __syncthreads();

  // ================= Phase 6: two GCN layers H = relu(A H W) + H =============
  for (int l = 0; l < 2; ++l) {
    const float* Wg = (l == 0) ? W0 : W1;
    for (int i = tid; i < 4096; i += 256) {
      int k = i >> 6, n = i & 63;
      wT72[n * 72 + k] = (_Float16)Wg[i];
    }
    __syncthreads();
    // T = A @ H  (K = 128 agents), store f16 into Y16 buffer
    {
      const _Float16* ARow = A16 + (m0 + l16) * 136;
      v8f acc[4] = {};
#pragma unroll
      for (int k0 = 0; k0 < 128; k0 += 32) {
        v16h a = ldA(ARow, k0, koffA);
#pragma unroll
        for (int nt = 0; nt < 4; ++nt)
          acc[nt] = wmma_f16(a, ldB(HT16 + (nt * 16 + l16) * 136, k0, koffB), acc[nt]);
      }
#pragma unroll
      for (int nt = 0; nt < 4; ++nt) {
        int n = nt * 16 + l16;
#pragma unroll
        for (int r = 0; r < 8; ++r)
          Y16[(m0 + r + half * 8) * 72 + n] = (_Float16)acc[nt][r];
      }
    }
    __syncthreads();
    // H = relu(T @ W) + H
    {
      const _Float16* Trow = Y16 + (m0 + l16) * 72;
      v8f acc[4] = {};
#pragma unroll
      for (int k0 = 0; k0 < 64; k0 += 32) {
        v16h a = ldA(Trow, k0, koffA);
#pragma unroll
        for (int nt = 0; nt < 4; ++nt)
          acc[nt] = wmma_f16(a, ldB(wT72 + (nt * 16 + l16) * 72, k0, koffB), acc[nt]);
      }
#pragma unroll
      for (int nt = 0; nt < 4; ++nt) {
        int n = nt * 16 + l16;
#pragma unroll
        for (int r = 0; r < 8; ++r) {
          int row = m0 + r + half * 8;
          float h = fmaxf(acc[nt][r], 0.f) + H32[row * 68 + n];
          H32[row * 68 + n] = h;
          HT16[n * 136 + row] = (_Float16)h;
        }
      }
    }
    __syncthreads();
  }

  // ================= Phase 7: write H to global ==============================
  for (int i = tid; i < 8192; i += 256) {
    int n = i >> 6, d = i & 63;
    out[(size_t)b * 8192 + i] = H32[n * 68 + d];
  }
}

extern "C" void kernel_launch(void* const* d_in, const int* in_sizes, int n_in,
                              void* d_out, int out_size, void* d_ws, size_t ws_size,
                              hipStream_t stream) {
  const float* robot = (const float*)d_in[0];
  const float* human = (const float*)d_in[1];
  const float* wr_w1 = (const float*)d_in[2];
  const float* wr_b1 = (const float*)d_in[3];
  const float* wr_w2 = (const float*)d_in[4];
  const float* wr_b2 = (const float*)d_in[5];
  const float* wh_w1 = (const float*)d_in[6];
  const float* wh_b1 = (const float*)d_in[7];
  const float* wh_w2 = (const float*)d_in[8];
  const float* wh_b2 = (const float*)d_in[9];
  const float* w_a   = (const float*)d_in[10];
  const float* wv_w  = (const float*)d_in[11];
  const float* wv_b  = (const float*)d_in[12];
  const float* W0    = (const float*)d_in[13];
  const float* W1    = (const float*)d_in[14];
  float* out = (float*)d_out;

  const int B = in_sizes[0] / 9;   // robot_state is [B,1,9]

  // Opt in to >64KB dynamic LDS (gfx1250 WGP has 320KB). Deterministic, not a
  // stream op, safe under graph capture; ignore failure.
  (void)hipFuncSetAttribute((const void*)gcn_fused_kernel,
                            hipFuncAttributeMaxDynamicSharedMemorySize,
                            (int)SMEM_BYTES);

  gcn_fused_kernel<<<dim3(B), dim3(256), SMEM_BYTES, stream>>>(
      robot, human, wr_w1, wr_b1, wr_w2, wr_b2,
      wh_w1, wh_b1, wh_w2, wh_b2, w_a, wv_w, wv_b, W0, W1, out);
}
